// AwqW4A8Linear_79963701117640
// MI455X (gfx1250) — 8-bit matrix operand order repaired and checked by reading, compiled, not run on hardware
//
#include <hip/hip_runtime.h>

typedef int          v8i   __attribute__((ext_vector_type(8)));
typedef unsigned int u32x4 __attribute__((ext_vector_type(4)));
typedef int          i32x8 __attribute__((ext_vector_type(8)));
typedef int          i32x4 __attribute__((ext_vector_type(4)));

#define M_DIM 256
#define K_DIM 8192
#define N_DIM 8192
#define BN    32
#define KSTEP 64
#define NIT   (K_DIM / KSTEP)
#define A_STRIDE 80
#define B_STRIDE 80

#define TDM_NONE 0
#if defined(__has_builtin)
#if __has_builtin(__builtin_amdgcn_tensor_load_to_lds)
#if __clang_major__ >= 23
#define USE_TDM 1
#define TDM_ARGS6 1
#else
#define USE_TDM 1
#define TDM_ARGS6 0
#endif
#endif
#endif
#ifndef USE_TDM
#define USE_TDM 0
#endif

__global__ __launch_bounds__(256) void quant_kernel(
    const float* __restrict__ x,
    signed char* __restrict__ xq,
    float* __restrict__ ascale)
{
    __shared__ float red[256];
    const int m = blockIdx.x;
    const int t = threadIdx.x;
    const float4* row = (const float4*)(x + (size_t)m * K_DIM);

    float4 v[8];
    float amax = 0.f;
#pragma unroll
    for (int i = 0; i < 8; ++i) {
        v[i] = row[i * 256 + t];
        amax = fmaxf(amax, fmaxf(fmaxf(fabsf(v[i].x), fabsf(v[i].y)),
                                 fmaxf(fabsf(v[i].z), fabsf(v[i].w))));
    }
    red[t] = amax;
    __syncthreads();
    for (int s = 128; s > 0; s >>= 1) {
        if (t < s) red[t] = fmaxf(red[t], red[t + s]);
        __syncthreads();
    }
    const float scale = fmaxf(red[0] * (1.0f / 127.0f), 1e-8f);
    const float inv   = 1.0f / scale;
    if (t == 0) ascale[m] = scale;

    unsigned int* out = (unsigned int*)(xq + (size_t)m * K_DIM);
#pragma unroll
    for (int i = 0; i < 8; ++i) {
        int q0 = (int)rintf(fminf(fmaxf(v[i].x * inv, -127.f), 127.f));
        int q1 = (int)rintf(fminf(fmaxf(v[i].y * inv, -127.f), 127.f));
        int q2 = (int)rintf(fminf(fmaxf(v[i].z * inv, -127.f), 127.f));
        int q3 = (int)rintf(fminf(fmaxf(v[i].w * inv, -127.f), 127.f));
        unsigned int packed = (q0 & 255) | ((q1 & 255) << 8) |
                              ((q2 & 255) << 16) | ((unsigned)(q3 & 255) << 24);
        out[i * 256 + t] = packed;
    }
}

#if USE_TDM
__device__ __forceinline__ void stage_a_tdm(const signed char* xq,
                                            signed char* dst, int k0) {
    unsigned long long ga = (unsigned long long)(uintptr_t)(xq + k0);
    unsigned lds_off = (unsigned)(uintptr_t)dst;
    u32x4 g0;
    g0[0] = 1u;
    g0[1] = lds_off;
    g0[2] = (unsigned)(ga & 0xffffffffull);
    g0[3] = (unsigned)((ga >> 32) & 0x1ffffffull)
          | (2u << 30);
    i32x8 g1;
    g1[0] = (1 << 20)
          | (3 << 22)
          | (3 << 25);
    g1[1] = (KSTEP & 0xffff) << 16;
    g1[2] = (M_DIM & 0xffff) << 16;
    g1[3] = (KSTEP << 16);
    g1[4] = M_DIM;
    g1[5] = K_DIM;
    g1[6] = 0;
    g1[7] = 0;
    i32x4 z4 = {0, 0, 0, 0};
#if TDM_ARGS6
    i32x8 z8 = {0, 0, 0, 0, 0, 0, 0, 0};
    __builtin_amdgcn_tensor_load_to_lds(g0, g1, z4, z4, z8, 0);
#else
    __builtin_amdgcn_tensor_load_to_lds(g0, g1, z4, z4, 0);
#endif
}
#endif

__global__ __launch_bounds__(256) void gemm_kernel(
    const signed char* __restrict__ xq,
    const int*         __restrict__ qw,
    const float*       __restrict__ ascale,
    const float*       __restrict__ wscale,
    const float*       __restrict__ bias,
    float*             __restrict__ out)
{
    __shared__ signed char lA[2][M_DIM * A_STRIDE];
    __shared__ signed char lB[2][BN * B_STRIDE];

    const int t    = threadIdx.x;
    const int wave = t >> 5;
    const int lane = t & 31;
    const int hf   = lane >> 4;
    const int l16  = lane & 15;
    const int n0   = blockIdx.x * BN;

    const int sb_n  = t & 31;
    const int sb_kq = t >> 5;

    v8i zero8 = 0;
    v8i acc00 = zero8, acc01 = zero8, acc10 = zero8, acc11 = zero8;

    int4 a0, a1, a2, a3;
    int  w0, w1, w2, w3, w4, w5, w6, w7;

#define LOAD_A(k0)                                                        \
    do {                                                                  \
        const int4* _p = (const int4*)(xq + (size_t)t * K_DIM + (k0));    \
        a0 = _p[0]; a1 = _p[1]; a2 = _p[2]; a3 = _p[3];                   \
    } while (0)
#define STORE_A(dst)                                                      \
    do {                                                                  \
        int4* _d = (int4*)((dst) + t * A_STRIDE);                         \
        _d[0] = a0; _d[1] = a1; _d[2] = a2; _d[3] = a3;                   \
    } while (0)
#define LOAD_W(k0)                                                        \
    do {                                                                  \
        const int* _p = qw + (size_t)((k0) + sb_kq * 8) * N_DIM + n0 + sb_n; \
        w0 = __builtin_nontemporal_load(_p + 0 * (size_t)N_DIM);          \
        w1 = __builtin_nontemporal_load(_p + 1 * (size_t)N_DIM);          \
        w2 = __builtin_nontemporal_load(_p + 2 * (size_t)N_DIM);          \
        w3 = __builtin_nontemporal_load(_p + 3 * (size_t)N_DIM);          \
        w4 = __builtin_nontemporal_load(_p + 4 * (size_t)N_DIM);          \
        w5 = __builtin_nontemporal_load(_p + 5 * (size_t)N_DIM);          \
        w6 = __builtin_nontemporal_load(_p + 6 * (size_t)N_DIM);          \
        w7 = __builtin_nontemporal_load(_p + 7 * (size_t)N_DIM);          \
    } while (0)
#define STORE_W(dst)                                                      \
    do {       \
        unsigned _t01 = __builtin_amdgcn_perm((unsigned)w1, (unsigned)w0, 0x00000400u); \
        unsigned _t23 = __builtin_amdgcn_perm((unsigned)w3, (unsigned)w2, 0x00000400u); \
        unsigned _t45 = __builtin_amdgcn_perm((unsigned)w5, (unsigned)w4, 0x00000400u); \
        unsigned _t67 = __builtin_amdgcn_perm((unsigned)w7, (unsigned)w6, 0x00000400u); \
        unsigned _d0  = __builtin_amdgcn_perm(_t23, _t01, 0x05040100u);   \
        unsigned _d1  = __builtin_amdgcn_perm(_t67, _t45, 0x05040100u);   \
        *(int2*)((dst) + sb_n * B_STRIDE + sb_kq * 8) = int2{(int)_d0, (int)_d1}; \
    } while (0)
#define COMPUTE(cA, cB)                                                   \
    do {                                                                  \
        const signed char* _bp0 = (cB) + (0 * 16 + l16) * B_STRIDE + hf * 16; \
        const signed char* _bp1 = (cB) + (1 * 16 + l16) * B_STRIDE + hf * 16; \
        int4 _b0l = *(const int4*)_bp0,  _b0h = *(const int4*)(_bp0 + 32); \
        int4 _b1l = *(const int4*)_bp1,  _b1h = *(const int4*)(_bp1 + 32); \
        v8i _bf0 = v8i{_b0l.x, _b0l.y, _b0l.z, _b0l.w, _b0h.x, _b0h.y, _b0h.z, _b0h.w}; \
        v8i _bf1 = v8i{_b1l.x, _b1l.y, _b1l.z, _b1l.w, _b1h.x, _b1h.y, _b1h.z, _b1h.w}; \
        const signed char* _ap0 = (cA) + ((wave * 2 + 0) * 16 + l16) * A_STRIDE + hf * 16; \
        const signed char* _ap1 = (cA) + ((wave * 2 + 1) * 16 + l16) * A_STRIDE + hf * 16; \
        int2 _p0 = *(const int2*)(_ap0 + 0),  _p1 = *(const int2*)(_ap0 + 8); \
        int2 _p2 = *(const int2*)(_ap0 + 32), _p3 = *(const int2*)(_ap0 + 40); \
        v8i _af0 = v8i{_p0.x, _p0.y, _p1.x, _p1.y, _p2.x, _p2.y, _p3.x, _p3.y}; \
        int2 _q0 = *(const int2*)(_ap1 + 0),  _q1 = *(const int2*)(_ap1 + 8); \
        int2 _q2 = *(const int2*)(_ap1 + 32), _q3 = *(const int2*)(_ap1 + 40); \
        v8i _af1 = v8i{_q0.x, _q0.y, _q1.x, _q1.y, _q2.x, _q2.y, _q3.x, _q3.y}; \
        acc00 = __builtin_amdgcn_wmma_i32_16x16x64_iu8(true, _af0, true, _bf0, acc00, false, false); \
        acc01 = __builtin_amdgcn_wmma_i32_16x16x64_iu8(true, _af0, true, _bf1, acc01, false, false); \
        acc10 = __builtin_amdgcn_wmma_i32_16x16x64_iu8(true, _af1, true, _bf0, acc10, false, false); \
        acc11 = __builtin_amdgcn_wmma_i32_16x16x64_iu8(true, _af1, true, _bf1, acc11, false, false); \
    } while (0)

#if USE_TDM
    if (t < 32) stage_a_tdm(xq, &lA[0][0], 0);
#else
    LOAD_A(0);
    STORE_A(&lA[0][0]);
#endif
    LOAD_W(0);
    STORE_W(&lB[0][0]);
#if USE_TDM
    if (t < 32) __builtin_amdgcn_s_wait_tensorcnt(0);
#endif
    __syncthreads();

    for (int it = 0; it < NIT; ++it) {
        const int cur = it & 1;
        const int kn  = (it + 1) * KSTEP;
        signed char* nA = &lA[cur ^ 1][0];
        signed char* nB = &lB[cur ^ 1][0];
        if (kn < K_DIM) {
#if USE_TDM
            if (t < 32) stage_a_tdm(xq, nA, kn);
#else
            LOAD_A(kn);
#endif
            LOAD_W(kn);
            __builtin_prefetch(
                qw + (size_t)(((kn + KSTEP) & (K_DIM - 1)) + sb_kq * 8) * N_DIM + n0 + sb_n,
                0, 1);
        }
        COMPUTE(&lA[cur][0], &lB[cur][0]);
        if (kn < K_DIM) {
#if !USE_TDM
            STORE_A(nA);
#endif
            STORE_W(nB);
#if USE_TDM
            if (t < 32) __builtin_amdgcn_s_wait_tensorcnt(0);
#endif
        }
        __syncthreads();
    }

#pragma unroll
    for (int mt = 0; mt < 2; ++mt) {
        const int mbase = (wave * 2 + mt) * 16 + hf * 8;
        const v8i acc_a = mt ? acc10 : acc00;
        const v8i acc_b = mt ? acc11 : acc01;
        float as0 = ascale[mbase + 0], as1 = ascale[mbase + 1];
        float as2 = ascale[mbase + 2], as3 = ascale[mbase + 3];
        float as4 = ascale[mbase + 4], as5 = ascale[mbase + 5];
        float as6 = ascale[mbase + 6], as7 = ascale[mbase + 7];
#pragma unroll
        for (int nt = 0; nt < 2; ++nt) {
            const v8i a = nt ? acc_b : acc_a;
            const int col = n0 + nt * 16 + l16;
            const float ws = wscale[col];
            const float bs = bias[col];
            float* op = out + (size_t)mbase * N_DIM + col;
            op[0 * N_DIM] = (float)a[0] * as0 * ws + bs;
            op[1 * N_DIM] = (float)a[1] * as1 * ws + bs;
            op[2 * N_DIM] = (float)a[2] * as2 * ws + bs;
            op[3 * N_DIM] = (float)a[3] * as3 * ws + bs;
            op[4 * N_DIM] = (float)a[4] * as4 * ws + bs;
            op[5 * N_DIM] = (float)a[5] * as5 * ws + bs;
            op[6 * N_DIM] = (float)a[6] * as6 * ws + bs;
            op[7 * N_DIM] = (float)a[7] * as7 * ws + bs;
        }
    }
#undef LOAD_A
#undef STORE_A
#undef LOAD_W
#undef STORE_W
#undef COMPUTE
}

extern "C" void kernel_launch(void* const* d_in, const int* in_sizes, int n_in,
                              void* d_out, int out_size, void* d_ws, size_t ws_size,
                              hipStream_t stream) {
    const float* x      = (const float*)d_in[0];
    const int*   qw     = (const int*)d_in[1];
    const float* wscale = (const float*)d_in[2];
    const float* bias   = (const float*)d_in[3];
    float*       out    = (float*)d_out;

    signed char* xq  = (signed char*)d_ws;
    float*       asc = (float*)((char*)d_ws + (size_t)M_DIM * K_DIM);

    quant_kernel<<<M_DIM, 256, 0, stream>>>(x, xq, asc);
    gemm_kernel<<<N_DIM / BN, 256, 0, stream>>>(xq, qw, asc, wscale, bias, out);
}
